// HieraBlock_57939108823144
// MI455X (gfx1250) — matrix operand order repaired and checked by reading, not compiled, not run on hardware
//
#include <hip/hip_runtime.h>
#include <hip/hip_bf16.h>
#include <math.h>

// ---------------------------------------------------------------------------
// Hiera block for MI455X (gfx1250): fused WMMA f16 pipeline, wave32.
// All operands pre-swizzled into per-lane WMMA fragment layout so every
// A/B fragment is ONE contiguous 32-byte load (2x b128), zero packing VALU.
// Cross-lane reductions use v_permlane16/v_permlanex16 (pure VALU, no LDS).
// ---------------------------------------------------------------------------

typedef _Float16 f16;
typedef f16  v16h __attribute__((ext_vector_type(16)));
typedef f16  v8h  __attribute__((ext_vector_type(8)));
typedef float v8f __attribute__((ext_vector_type(8)));

#define DEV __device__ __forceinline__

DEV v8f wmma16(v16h a, v16h b, v8f c) {
  return __builtin_amdgcn_wmma_f32_16x16x32_f16(false, a, false, b, (short)0, c,
                                                false, false);
}

// One 32-byte fragment load from packed storage.
DEV v16h frag(const f16* p, int tile, int lane) {
  return *(const v16h*)(p + ((tile * 32 + lane) << 4));
}

// A-fragment register order: k-offset r (0..31) -> (exec-half, element idx).
DEV void a_coord(int r, int& hi_a, int& i) {
  hi_a = r >> 4; i = r & 15;
}

// ---- cross-lane helpers: VALU permlanes (fallback: shfl via LDS) ----------
#if __has_builtin(__builtin_amdgcn_permlanex16) && __has_builtin(__builtin_amdgcn_permlane16)
#define HAVE_PERMLANE 1
#endif

DEV float xhalf_swap(float v) {   // lane <- lane ^ 16
#ifdef HAVE_PERMLANE
  int iv = __float_as_int(v);
  iv = __builtin_amdgcn_permlanex16(iv, iv, (int)0x76543210, (int)0xfedcba98,
                                    false, false);
  return __int_as_float(iv);
#else
  return __shfl_xor(v, 16, 32);
#endif
}

#ifdef HAVE_PERMLANE
DEV float pl16(float v, int s0, int s1) {  // lane <- group16[sel]
  int iv = __float_as_int(v);
  iv = __builtin_amdgcn_permlane16(iv, iv, s0, s1, false, false);
  return __int_as_float(iv);
}
#endif

DEV float wsum(float v) {
#ifdef HAVE_PERMLANE
  v += xhalf_swap(v);                                     // xor 16
  v += pl16(v, (int)0xfedcba98, (int)0x76543210);         // xor 8
  v += pl16(v, (int)0x32107654, (int)0xba98fedc);         // xor 4
  v += pl16(v, (int)0x54761032, (int)0xdcfe98ba);         // xor 2
  v += pl16(v, (int)0x67452301, (int)0xefcdab89);         // xor 1
  return v;
#else
#pragma unroll
  for (int m = 16; m >= 1; m >>= 1) v += __shfl_xor(v, m, 32);
  return v;
#endif
}

// ---------------------------------------------------------------------------
// k0: fp32 weights -> f16 packed B-fragment layout in workspace.
// ws f16 offsets: qkv 0 | skip 55296 | proj 73728 | w1 110592 | w2 258048
// ---------------------------------------------------------------------------
DEV void pack_one(const float* __restrict__ src, f16* __restrict__ dst,
                  int NT, int N, int p) {
  int ii = p & 15;
  int lane = (p >> 4) & 31;
  int t = p >> 9;
  int nt = t % NT, kt = t / NT;
  int row = kt * 32 + (lane >> 4) * 16 + ii;   // K index
  int col = nt * 16 + (lane & 15);             // N index
  dst[p] = (f16)src[row * N + col];
}

__global__ void k0_pack(const float* __restrict__ qkv, const float* __restrict__ skip,
                        const float* __restrict__ proj, const float* __restrict__ w1,
                        const float* __restrict__ w2, f16* __restrict__ ws) {
  int i = blockIdx.x * 256 + threadIdx.x;
  if (i < 55296)        pack_one(qkv,  ws,          36, 576, i);
  else if (i < 73728)   pack_one(skip, ws + 55296,  12, 192, i - 55296);
  else if (i < 110592)  pack_one(proj, ws + 73728,  12, 192, i - 73728);
  else if (i < 258048)  pack_one(w1,   ws + 110592, 48, 768, i - 110592);
  else if (i < 405504)  pack_one(w2,   ws + 258048, 12, 192, i - 258048);
}

// ---------------------------------------------------------------------------
// k1: LN1 over DIN=96 (one wave per token), f16 cast, window partition,
// output directly in packed A-fragment layout: xn[win][tile=mt*3+kt][lane][16]
// ---------------------------------------------------------------------------
__global__ __launch_bounds__(256) void k1_ln1(const float* __restrict__ x,
                                              const float* __restrict__ g,
                                              const float* __restrict__ bb,
                                              f16* __restrict__ xn) {
  const int tid = threadIdx.x, lane = tid & 31;
  const int token = (blockIdx.x * 256 + tid) >> 5;  // 131072 tokens
  const float* px = x + (size_t)token * 96;
  float v0 = px[lane], v1 = px[lane + 32], v2 = px[lane + 64];
  float s  = wsum(v0 + v1 + v2);
  float ss = wsum(v0 * v0 + v1 * v1 + v2 * v2);
  float mean = s * (1.f / 96.f);
  float var  = ss * (1.f / 96.f) - mean * mean;
  float rs   = rsqrtf(var + 1e-6f);
  const int b = token >> 16, rem = token & 0xFFFF;
  const int hh = rem >> 8, wp = rem & 0xFF;
  const int win = b * 1024 + (hh >> 3) * 32 + (wp >> 3);
  const int tok = (hh & 7) * 8 + (wp & 7);
  const int mt = tok >> 4, lmt = tok & 15;
  f16* pw = xn + (size_t)win * 6144;   // 64*96 halfs per window
  float vv[3] = {v0, v1, v2};
#pragma unroll
  for (int q = 0; q < 3; ++q) {
    int c = lane + 32 * q;
    float val = (vv[q] - mean) * rs * g[c] + bb[c];
    int kt = c >> 5, r = c & 31, hi_a, i;
    a_coord(r, hi_a, i);
    pw[((mt * 3 + kt) * 32 + lmt + 16 * hi_a) * 16 + i] = (f16)val;
  }
}

// ---------------------------------------------------------------------------
// k2: per-window fused attention + skip path (one workgroup = one 8x8 window,
// 8 wave32s). Window activations staged LDS via async global->LDS copy.
// ---------------------------------------------------------------------------
__global__ __launch_bounds__(256) void k2_attn(
    const f16* __restrict__ xn,       // packed A frags [2048][12][32][16]
    const f16* __restrict__ Wqkv,     // packed B frags, KT=3 NT=36
    const float* __restrict__ bqkv,   // [576]
    const f16* __restrict__ Wskip,    // packed B frags, KT=3 NT=12
    const float* __restrict__ bskip,  // [192]
    const f16* __restrict__ Wproj,    // packed B frags, KT=6 NT=12
    const float* __restrict__ bproj,  // [192]
    float* __restrict__ out)          // y: [2][128][128][192]
{
  __shared__ __align__(32) f16   sA[12 * 512];    // activations, A frags
  __shared__ __align__(32) float sSkip[16 * 192]; // pooled skip (f32)
  __shared__ __align__(32) f16   sKT[12 * 512];   // K^T, B frags (KT=3,NT=4)
  __shared__ __align__(32) f16   sV[12 * 512];    // V, B frags (KT=2,NT=6)
  __shared__ __align__(32) f16   sQP[3 * 512];    // pooled Q, A frags (KT=3)
  __shared__ __align__(32) float sS[16 * 64];     // scores (f32)
  __shared__ __align__(32) f16   sAt[2 * 512];    // attn probs, A frags (KT=2)
  __shared__ __align__(32) f16   sO[6 * 512];     // attn out, A frags (KT=6)

  const int tid = threadIdx.x;
  const int wave = tid >> 5, lane = tid & 31, lm = lane & 15, hi = lane >> 4;
  const int win = blockIdx.x;
  const int b = win >> 10, wh = (win >> 5) & 31, ww = win & 31;

  // ---- async stage of this window's activations (12288 B = 768 x b128).
  {
    const f16* src = xn + (size_t)win * 6144;
    unsigned lds_base = (unsigned)(uintptr_t)(&sA[0]);
#pragma unroll
    for (int c = 0; c < 3; ++c) {
      unsigned chunk = tid + 256 * c;                // 0..767
      unsigned loff = lds_base + chunk * 16;
      unsigned goff = chunk * 16;
      asm volatile("global_load_async_to_lds_b128 %0, %1, %2"
                   :: "v"(loff), "v"(goff), "s"(src)
                   : "memory");
    }
#if __has_builtin(__builtin_amdgcn_s_wait_asynccnt)
    __builtin_amdgcn_s_wait_asynccnt(0);
#else
    asm volatile("s_wait_asynccnt 0x0" ::: "memory");
#endif
  }
  __syncthreads();

  // ---- skip: (64x96)@(96x192) with fused 2x2 maxpool -> sSkip[16][192].
  // Both EXEC halves end with the identical pooled value -> unconditional
  // same-address store (no exec-mask branches).
  for (int tt = wave; tt < 48; tt += 8) {
    int mt = tt / 12, nt = tt % 12;
    v8f acc = {0.f, 0.f, 0.f, 0.f, 0.f, 0.f, 0.f, 0.f};
#pragma unroll
    for (int ks = 0; ks < 3; ++ks)
      acc = wmma16(frag(sA, mt * 3 + ks, lane), frag(Wskip, ks * 12 + nt, lane), acc);
    float bias = bskip[nt * 16 + lm];
#pragma unroll
    for (int C = 0; C < 4; ++C) {
      float p = fmaxf(acc[2 * C], acc[2 * C + 1]);
      p = fmaxf(p, xhalf_swap(p));
      sSkip[(mt * 4 + C) * 192 + nt * 16 + lm] = p + bias;
    }
  }

  const float scale = 0.1020620726159658f;  // 96^-0.5
#pragma unroll 1
  for (int h = 0; h < 2; ++h) {
    __syncthreads();
    // K: (64x96)@(96x96) -> B-fragment layout for scores (KT=3 over d, NT=4 over tok)
    for (int tt = wave; tt < 24; tt += 8) {
      int mt = tt / 6, nt6 = tt % 6;
      v8f acc = {0.f, 0.f, 0.f, 0.f, 0.f, 0.f, 0.f, 0.f};
#pragma unroll
      for (int ks = 0; ks < 3; ++ks)
        acc = wmma16(frag(sA, mt * 3 + ks, lane),
                     frag(Wqkv, ks * 36 + 12 + h * 6 + nt6, lane), acc);
      float bias = bqkv[192 + h * 96 + nt6 * 16 + lm];
      // element (d = nt6*16+lm, tok = mt*16+8hi+j) -> B frag [d/32][tok/16]
      int base = (((nt6 >> 1) * 4 + mt) * 32 + 16 * (nt6 & 1)) * 16 + lm;
#pragma unroll
      for (int j = 0; j < 8; ++j)
        sKT[base + (8 * hi + j) * 16] = (f16)(acc[j] + bias);
    }
    // V: (64x96)@(96x96) -> B-fragment layout for attn@V (KT=2 over tok, NT=6 over d)
    for (int tt = wave; tt < 24; tt += 8) {
      int mt = tt / 6, nt6 = tt % 6;
      v8f acc = {0.f, 0.f, 0.f, 0.f, 0.f, 0.f, 0.f, 0.f};
#pragma unroll
      for (int ks = 0; ks < 3; ++ks)
        acc = wmma16(frag(sA, mt * 3 + ks, lane),
                     frag(Wqkv, ks * 36 + 24 + h * 6 + nt6, lane), acc);
      float bias = bqkv[384 + h * 96 + nt6 * 16 + lm];
      v8h pk;
#pragma unroll
      for (int j = 0; j < 8; ++j) pk[j] = (f16)(acc[j] + bias);
      // contiguous: element i = 8hi..8hi+7, lane_b = 16*(mt&1)+lm, tile ks=mt>>1
      *(v8h*)(&sV[((((mt >> 1) * 6 + nt6) * 32 + 16 * (mt & 1) + lm) << 4) + 8 * hi]) = pk;
    }
    // Q with fused 2x2 maxpool -> A-fragment layout sQP (M=16 pooled, KT=3)
    for (int tt = wave; tt < 24; tt += 8) {
      int mt = tt / 6, nt6 = tt % 6;
      v8f acc = {0.f, 0.f, 0.f, 0.f, 0.f, 0.f, 0.f, 0.f};
#pragma unroll
      for (int ks = 0; ks < 3; ++ks)
        acc = wmma16(frag(sA, mt * 3 + ks, lane),
                     frag(Wqkv, ks * 36 + h * 6 + nt6, lane), acc);
      float bias = bqkv[h * 96 + nt6 * 16 + lm];
      int c = nt6 * 16 + lm, kt = c >> 5, r = c & 31, hia, ii;
      a_coord(r, hia, ii);
#pragma unroll
      for (int C = 0; C < 4; ++C) {
        float p = fmaxf(acc[2 * C], acc[2 * C + 1]);
        p = fmaxf(p, xhalf_swap(p));
        sQP[(kt * 32 + (mt * 4 + C) + 16 * hia) * 16 + ii] = (f16)(p + bias);
      }
    }
    __syncthreads();
    // scores: (16x96)@(96x64) * scale.
    if (wave < 4) {
      int nt = wave;
      v8f acc = {0.f, 0.f, 0.f, 0.f, 0.f, 0.f, 0.f, 0.f};
#pragma unroll
      for (int ks = 0; ks < 3; ++ks)
        acc = wmma16(frag(sQP, ks, lane), frag(sKT, ks * 4 + nt, lane), acc);
#pragma unroll
      for (int j = 0; j < 8; ++j)
        sS[(8 * hi + j) * 64 + nt * 16 + lm] = acc[j] * scale;
    }
    __syncthreads();
    // softmax over 64 keys; write probs in A-fragment layout (KT=2).
    if (tid < 16) {
      float mx = -1e30f;
      for (int s = 0; s < 64; ++s) mx = fmaxf(mx, sS[tid * 64 + s]);
      float sum = 0.f;
      for (int s = 0; s < 64; ++s) {
        float e = expf(sS[tid * 64 + s] - mx);
        sS[tid * 64 + s] = e;
        sum += e;
      }
      float inv = 1.f / sum;
      for (int s = 0; s < 64; ++s) {
        int kt = s >> 5, r = s & 31, hia, ii;
        a_coord(r, hia, ii);
        sAt[(kt * 32 + tid + 16 * hia) * 16 + ii] = (f16)(sS[tid * 64 + s] * inv);
      }
    }
    __syncthreads();
    // out = attn (16x64) @ V (64x96) -> A-fragment layout sO (KT=6 over 192).
    if (wave < 6) {
      int nt = wave;
      v8f acc = {0.f, 0.f, 0.f, 0.f, 0.f, 0.f, 0.f, 0.f};
#pragma unroll
      for (int ks = 0; ks < 2; ++ks)
        acc = wmma16(frag(sAt, ks, lane), frag(sV, ks * 6 + nt, lane), acc);
      int c = h * 96 + nt * 16 + lm, kt = c >> 5, r = c & 31, hia, ii;
      a_coord(r, hia, ii);
#pragma unroll
      for (int j = 0; j < 8; ++j)
        sO[(kt * 32 + (8 * hi + j) + 16 * hia) * 16 + ii] = (f16)acc[j];
    }
    __syncthreads();  // protect sKT/sV/sS/sAt reuse across heads
  }

  // ---- proj: (16x192)@(192x192) + bias + pooled skip -> y (d_out).
  for (int nt = wave; nt < 12; nt += 8) {
    v8f acc = {0.f, 0.f, 0.f, 0.f, 0.f, 0.f, 0.f, 0.f};
#pragma unroll
    for (int ks = 0; ks < 6; ++ks)
      acc = wmma16(frag(sO, ks, lane), frag(Wproj, ks * 12 + nt, lane), acc);
    float bias = bproj[nt * 16 + lm];
#pragma unroll
    for (int j = 0; j < 8; ++j) {
      int m = 8 * hi + j, R = m >> 2, C = m & 3;
      float v = acc[j] + bias + sSkip[m * 192 + nt * 16 + lm];
      out[(((size_t)b * 128 + wh * 4 + R) * 128 + ww * 4 + C) * 192 + nt * 16 + lm] = v;
    }
  }
}

// ---------------------------------------------------------------------------
// k3: LN2 + MLP (192 -> 768 -> 192, exact-erf GELU) + residual into d_out.
// One workgroup per 16 pooled tokens; LDS intermediates in A-fragment layout.
// ---------------------------------------------------------------------------
__global__ __launch_bounds__(256) void k3_mlp(
    const float* __restrict__ g2, const float* __restrict__ b2,
    const f16* __restrict__ W1, const float* __restrict__ b1,
    const f16* __restrict__ W2, const float* __restrict__ bb2,
    float* __restrict__ out) {
  __shared__ __align__(32) f16 sY[6 * 512];    // LN2 out, A frags (KT=6)
  __shared__ __align__(32) f16 sH[24 * 512];   // GELU(h),  A frags (KT=24)
  const int tid = threadIdx.x, wave = tid >> 5, lane = tid & 31;
  const int lm = lane & 15, hi = lane >> 4;
  const size_t t0 = (size_t)blockIdx.x * 16;

  // LN2: each wave handles 2 tokens, lane-parallel over 192 channels.
#pragma unroll 1
  for (int tt = wave * 2; tt < wave * 2 + 2; ++tt) {
    const float* y = out + (t0 + tt) * 192;
    float v[6], s = 0.f, ss = 0.f;
#pragma unroll
    for (int q = 0; q < 6; ++q) {
      v[q] = y[lane + 32 * q];
      s += v[q];
      ss += v[q] * v[q];
    }
    s = wsum(s);
    ss = wsum(ss);
    float mean = s * (1.f / 192.f);
    float var = ss * (1.f / 192.f) - mean * mean;
    float rs = rsqrtf(var + 1e-6f);
#pragma unroll
    for (int q = 0; q < 6; ++q) {
      int ci = lane + 32 * q;
      float val = (v[q] - mean) * rs * g2[ci] + b2[ci];
      int kt = ci >> 5, r = ci & 31, hia, ii;
      a_coord(r, hia, ii);
      sY[(kt * 32 + tt + 16 * hia) * 16 + ii] = (f16)val;
    }
  }
  __syncthreads();

  // GEMM1: (16x192)@(192x768) + b1, exact GELU -> sH (A frags).
  for (int nt = wave; nt < 48; nt += 8) {
    v8f acc = {0.f, 0.f, 0.f, 0.f, 0.f, 0.f, 0.f, 0.f};
#pragma unroll
    for (int ks = 0; ks < 6; ++ks)
      acc = wmma16(frag(sY, ks, lane), frag(W1, ks * 48 + nt, lane), acc);
    float bias = b1[nt * 16 + lm];
    int c = nt * 16 + lm, kt = c >> 5, r = c & 31, hia, ii;
    a_coord(r, hia, ii);
#pragma unroll
    for (int j = 0; j < 8; ++j) {
      float xv = acc[j] + bias;
      float ge = 0.5f * xv * (1.f + erff(xv * 0.7071067811865476f));
      sH[(kt * 32 + (8 * hi + j) + 16 * hia) * 16 + ii] = (f16)ge;
    }
  }
  __syncthreads();

  // GEMM2: (16x768)@(768x192) + b2 + residual(y) -> out.
  for (int nt = wave; nt < 12; nt += 8) {
    v8f acc = {0.f, 0.f, 0.f, 0.f, 0.f, 0.f, 0.f, 0.f};
#pragma unroll
    for (int ks = 0; ks < 24; ++ks)
      acc = wmma16(frag(sH, ks, lane), frag(W2, ks * 12 + nt, lane), acc);
    float bias = bb2[nt * 16 + lm];
#pragma unroll
    for (int j = 0; j < 8; ++j) {
      size_t idx = (t0 + 8 * hi + j) * 192 + nt * 16 + lm;
      out[idx] = out[idx] + acc[j] + bias;
    }
  }
}

// ---------------------------------------------------------------------------
extern "C" void kernel_launch(void* const* d_in, const int* in_sizes, int n_in,
                              void* d_out, int out_size, void* d_ws, size_t ws_size,
                              hipStream_t stream) {
  const float* x     = (const float*)d_in[0];
  const float* g1    = (const float*)d_in[1];
  const float* b1n   = (const float*)d_in[2];
  const float* qkvw  = (const float*)d_in[3];
  const float* qkvb  = (const float*)d_in[4];
  const float* projw = (const float*)d_in[5];
  const float* projb = (const float*)d_in[6];
  const float* skw   = (const float*)d_in[7];
  const float* skb   = (const float*)d_in[8];
  const float* g2    = (const float*)d_in[9];
  const float* b2    = (const float*)d_in[10];
  const float* w1    = (const float*)d_in[11];
  const float* mb1   = (const float*)d_in[12];
  const float* w2    = (const float*)d_in[13];
  const float* mb2   = (const float*)d_in[14];
  float* out = (float*)d_out;

  f16* ws    = (f16*)d_ws;
  f16* qkvh  = ws;
  f16* skiph = ws + 55296;
  f16* projh = ws + 73728;
  f16* w1h   = ws + 110592;
  f16* w2h   = ws + 258048;
  f16* xnh   = ws + 405504;  // packed A frags: [2048][12][32][16] f16

  k0_pack<<<dim3(1584), dim3(256), 0, stream>>>(qkvw, skw, projw, w1, w2, ws);
  k1_ln1<<<dim3(16384), dim3(256), 0, stream>>>(x, g1, b1n, xnh);
  k2_attn<<<dim3(2048), dim3(256), 0, stream>>>(xnh, qkvh, qkvb, skiph, skb,
                                                projh, projb, out);
  k3_mlp<<<dim3(2048), dim3(256), 0, stream>>>(g2, b2, w1h, mb1, w2h, mb2, out);
  (void)in_sizes; (void)n_in; (void)out_size; (void)ws_size;
}
